// GCN_alignment_cam_47502338294132
// MI455X (gfx1250) — matrix operand order repaired and checked by reading, not compiled, not run on hardware
//
#include <hip/hip_runtime.h>

// ---------------------------------------------------------------------------
// Problem: N=8192 nodes, H=256 hidden, B=256 batch, L=128 words.
//  K1: layer1 = A[8192,8192] @ E[8192,256]   (memory-bound: 272MB @ 23.3TB/s)
//      -> bf16 hi/lo split WMMA (3 products) = fp32-class accuracy for free.
//      B tiles streamed into double-buffered LDS by the Tensor Data Mover
//      (tensor_load_to_lds + s_wait_tensorcnt), overlapped with WMMA.
//  K2: fused per-batch attention+reduction+logits in one WG (213KB LDS).
// ---------------------------------------------------------------------------

typedef __attribute__((ext_vector_type(16))) __bf16         v16bf;
typedef __attribute__((ext_vector_type(8)))  float          v8f;
typedef __attribute__((ext_vector_type(4)))  float          f32x4;
typedef __attribute__((ext_vector_type(16))) unsigned short u16x16;
typedef __attribute__((ext_vector_type(8)))  unsigned short u16x8;

union PU16 { unsigned int w[8]; u16x16 v; };
union PU8  { unsigned int w[4]; u16x8  v; };

static __device__ __forceinline__ unsigned short f2bf(float f) {
  unsigned int u = __float_as_uint(f);
  u += 0x7fffu + ((u >> 16) & 1u);          // round-to-nearest-even
  return (unsigned short)(u >> 16);
}
static __device__ __forceinline__ float bf2f(unsigned short h) {
  return __uint_as_float(((unsigned int)h) << 16);
}
static __device__ __forceinline__ v8f v8zero() {
  v8f z = {0.f, 0.f, 0.f, 0.f, 0.f, 0.f, 0.f, 0.f};
  return z;
}
static __device__ __forceinline__ v8f wmma_bf16(u16x16 a, u16x16 b, v8f c) {
  union { u16x16 u; v16bf b; } ua, ub;
  ua.u = a; ub.u = b;
  // (neg_a, A, neg_b, B, c_mod, C, reuse_a, reuse_b)
  return __builtin_amdgcn_wmma_f32_16x16x32_bf16(false, ua.b, false, ub.b,
                                                 (short)0, c, false, false);
}

#if __has_builtin(__builtin_amdgcn_tensor_load_to_lds) && \
    __has_builtin(__builtin_amdgcn_s_wait_tensorcnt)
#define HAVE_TDM 1
#else
#define HAVE_TDM 0
#endif

#if HAVE_TDM
typedef __attribute__((ext_vector_type(4))) unsigned int u32x4;
typedef __attribute__((ext_vector_type(8))) int          i32x8;
typedef __attribute__((ext_vector_type(4))) int          i32x4;

// TDM descriptor for one K-tile of the pre-split embedding:
// 3D tile: X = 32 u16 (one 64B k-run), Y = 256 rows, Z = 2 planes (hi, lo).
// LDS padding: 16 DWORDs data then 4 DWORDs pad -> 80B row stride (bank-spread),
// matching the compute-side [n][k] stride-40-u16 layout, planes contiguous.
static __device__ __forceinline__ void tdm_issue_tile(const unsigned short* gtile,
                                                      unsigned lds_byte) {
  const unsigned long long ga = (unsigned long long)(size_t)gtile;
  u32x4 g0;
  g0[0] = 1u;                                              // count=1 (valid, user)
  g0[1] = lds_byte;                                        // lds_addr
  g0[2] = (unsigned)ga;                                    // global_addr[31:0]
  g0[3] = (unsigned)((ga >> 32) & 0x01ffffffu) | (2u << 30); // ga[56:32] | type=2
  i32x8 g1;
  g1[0] = (1 << 16)        // data_size = 1 -> 2 bytes
        | (1 << 20)        // pad_enable
        | (3 << 22)        // pad_interval: 16 DWORDs (64B of data)
        | (3 << 25);       // pad_amount:    4 DWORDs (16B of pad)
  g1[1] = (int)(0x2000u << 16);     // tensor_dim0 = 8192 (lo16 at bits 63:48)
  g1[2] = (int)(0x0100u << 16);     // tensor_dim0 hi16 = 0 | tensor_dim1 = 256
  g1[3] = (int)(32u << 16);         // tensor_dim1 hi16 = 0 | tile_dim0 = 32
  g1[4] = (int)(256u | (2u << 16)); // tile_dim1 = 256 | tile_dim2 = 2
  g1[5] = (int)8192;                // tensor_dim0_stride[31:0] = 8192
  g1[6] = 0;                        // stride0 hi | tensor_dim1_stride lo16
  g1[7] = 0x20;                     // tensor_dim1_stride[47:16]: 0x200000>>16 (plane)
  i32x4 g2;
  g2[0] = 2;                        // tensor_dim2 = 2 (planes)
  g2[1] = 0; g2[2] = 0; g2[3] = 0;
  i32x4 g3 = {0, 0, 0, 0};
  i32x8 g4 = {0, 0, 0, 0, 0, 0, 0, 0};   // unused trailing group (6-arg form)
  __builtin_amdgcn_tensor_load_to_lds(g0, g1, g2, g3, g4, 0);
}
#endif

// ---------------------------------------------------------------------------
// Kernel 0: split embedding E[8192,256] f32 into transposed, K-contiguous
// bf16 planes Et[2][256][8192] (hi plane then lo plane). 8MB -> L2-resident.
// ---------------------------------------------------------------------------
__global__ __launch_bounds__(256) void k_prep_E(const float* __restrict__ E,
                                                unsigned short* __restrict__ Ethi,
                                                unsigned short* __restrict__ Etlo) {
  const int id = blockIdx.x * 256 + threadIdx.x;   // 0 .. 262143
  const int n  = id & 255;
  const int k0 = (id >> 8) * 8;                    // 0 .. 8184
  PU8 hp, lp;
#pragma unroll
  for (int i = 0; i < 8; i += 2) {
    float v0 = E[(size_t)(k0 + i)     * 256 + n];
    float v1 = E[(size_t)(k0 + i + 1) * 256 + n];
    unsigned short h0 = f2bf(v0), h1 = f2bf(v1);
    unsigned short g0 = f2bf(v0 - bf2f(h0));
    unsigned short g1 = f2bf(v1 - bf2f(h1));
    hp.w[i >> 1] = (unsigned int)h0 | ((unsigned int)h1 << 16);
    lp.w[i >> 1] = (unsigned int)g0 | ((unsigned int)g1 << 16);
  }
  *(u16x8*)(&Ethi[(size_t)n * 8192 + k0]) = hp.v;
  *(u16x8*)(&Etlo[(size_t)n * 8192 + k0]) = lp.v;
}

// ---------------------------------------------------------------------------
// Kernel 1: layer1[8192,256] = A @ E.
// grid 128 x 128 threads (4 waves). Wave owns 16 rows x 256 cols
// (16 accumulator tiles) so A is streamed exactly once. K-step = 32.
// LDS: two buffers of [2 planes][256][40 u16] = 80KB, TDM double-buffered.
// ---------------------------------------------------------------------------
#define K1_BUF_U16   20480               // u16 per buffer (2 planes * 256*40)
#define K1_PLANE_U16 10240               // u16 per plane
#define K1_LDS_BYTES (2 * K1_BUF_U16 * 2)

__global__ __launch_bounds__(128) void k_gemm_layer1(
    const float* __restrict__ A,
    const unsigned short* __restrict__ Et,   // [2][256][8192] hi,lo planes
    float* __restrict__ C) {
  extern __shared__ __align__(16) char smem[];
  unsigned short* sB = (unsigned short*)smem;

  const int t    = threadIdx.x;
  const int wave = t >> 5;
  const int lane = t & 31;
  const int m    = lane & 15;   // row within tile (A) / col within tile (B,C)
  const int hh   = lane >> 4;   // lane half
  const int rowbase = blockIdx.x * 64 + wave * 16;
  const float* arow = A + (size_t)(rowbase + m) * 8192;

  v8f acc[16];
#pragma unroll
  for (int i = 0; i < 16; ++i) acc[i] = v8zero();

#if HAVE_TDM
  if (wave == 0) tdm_issue_tile(Et, (unsigned)(size_t)sB);
#endif

  for (int kk = 0; kk < 256; ++kk) {
    const int k0 = kk * 32;
#if HAVE_TDM
    const unsigned short* bhBase = sB + (kk & 1) * K1_BUF_U16;
    if (wave == 0) {
      if (kk + 1 < 256) {
        tdm_issue_tile(Et + (size_t)k0 + 32,
                       (unsigned)(size_t)(sB + ((kk + 1) & 1) * K1_BUF_U16));
        __builtin_amdgcn_s_wait_tensorcnt(1);   // current tile complete
      } else {
        __builtin_amdgcn_s_wait_tensorcnt(0);   // last tile complete
      }
    }
    __syncthreads();                            // tile visible to all waves
#else
    const unsigned short* bhBase = sB;
    __syncthreads();
    for (int nn = t; nn < 256; nn += 128) {     // fallback cooperative stage
      const u16x16* sh = (const u16x16*)(&Et[(size_t)nn * 8192 + k0]);
      const u16x16* sl = (const u16x16*)(&Et[2097152u + (size_t)nn * 8192 + k0]);
      *(u16x16*)(&sB[nn * 40 + 0])                 = sh[0];
      *(u16x16*)(&sB[nn * 40 + 16])                = sh[1];
      *(u16x16*)(&sB[K1_PLANE_U16 + nn * 40 + 0])  = sl[0];
      *(u16x16*)(&sB[K1_PLANE_U16 + nn * 40 + 16]) = sl[1];
    }
    __syncthreads();
#endif

    // --- A fragment: lane half hh needs K run [16h..16h+16) ---
    f32x4 q0 = *(const f32x4*)(arow + k0 + 16 * hh);
    f32x4 q1 = *(const f32x4*)(arow + k0 + 16 * hh + 4);
    f32x4 q2 = *(const f32x4*)(arow + k0 + 16 * hh + 8);
    f32x4 q3 = *(const f32x4*)(arow + k0 + 16 * hh + 12);
    __builtin_prefetch(arow + k0 + 512, 0, 1);   // global_prefetch ahead on A
    float av[16];
#pragma unroll
    for (int i = 0; i < 4; ++i) {
      av[i] = q0[i]; av[4 + i] = q1[i]; av[8 + i] = q2[i]; av[12 + i] = q3[i];
    }
    u16x16 ah, al;
#pragma unroll
    for (int i = 0; i < 16; ++i) {
      unsigned short x = f2bf(av[i]);
      ah[i] = x;
      al[i] = f2bf(av[i] - bf2f(x));
    }

#pragma unroll
    for (int tj = 0; tj < 16; ++tj) {
      u16x16 bh = *(const u16x16*)(&bhBase[(tj * 16 + m) * 40 + 16 * hh]);
      u16x16 bl = *(const u16x16*)(&bhBase[K1_PLANE_U16 + (tj * 16 + m) * 40 + 16 * hh]);
      acc[tj] = wmma_bf16(ah, bh, acc[tj]);   // hi*hi
      acc[tj] = wmma_bf16(ah, bl, acc[tj]);   // hi*lo
      acc[tj] = wmma_bf16(al, bh, acc[tj]);   // lo*hi
    }
#if HAVE_TDM
    __syncthreads();   // all waves done with this buffer before TDM reuses it
#endif
  }

  // --- write C: lane half hh -> rows v + 8*hh, col = tj*16 + m ---
#pragma unroll
  for (int tj = 0; tj < 16; ++tj) {
    const int col = tj * 16 + m;
#pragma unroll
    for (int v = 0; v < 8; ++v) {
      const int row = rowbase + v + 8 * hh;
      C[(size_t)row * 256 + col] = acc[tj][v];
    }
  }
}

// ---------------------------------------------------------------------------
// Kernel 2: fused per-batch attention. One WG (256 thr, 8 waves) per batch b.
// LDS map (213 KB of the 320 KB WGP budget):
//   R1 [69632B]: LQ u16[128][264]   -> later LQT u16[256][136]
//   R2 [69632B]: LA u16[128][264]   -> later LAT u16[256][136]
//   R3 [69632B]: S  f32[128][132]   -> later WQ u16[128][136] + WAT u16[128][136]
//   stats 2048B, contrib 2048B
// ---------------------------------------------------------------------------
#define SM_R1      0
#define SM_R2      69632
#define SM_R3      139264
#define SM_STATS   208896
#define SM_CONTRIB 210944
#define SM_TOTAL   212992

__global__ __launch_bounds__(256) void k_attn(
    const float* __restrict__ layer1,
    const float* __restrict__ Wmat,    // [2][1024]
    const float* __restrict__ bias,    // [2]
    const int*   __restrict__ qidx,
    const int*   __restrict__ aidx,
    float* __restrict__ out) {
  extern __shared__ __align__(16) char smem[];
  unsigned short* R1u  = (unsigned short*)(smem + SM_R1);
  unsigned short* R2u  = (unsigned short*)(smem + SM_R2);
  float*          S    = (float*)(smem + SM_R3);
  unsigned short* WQp  = (unsigned short*)(smem + SM_R3);
  unsigned short* WATp = (unsigned short*)(smem + SM_R3 + 34816);
  float* rowmax  = (float*)(smem + SM_STATS);
  float* rowsum  = rowmax + 128;
  float* colmax  = rowsum + 128;
  float* colsum  = colmax + 128;
  float* contrib = (float*)(smem + SM_CONTRIB);

  const int b    = blockIdx.x;
  const int t    = threadIdx.x;
  const int w    = t >> 5;
  const int lane = t & 31;
  const int ln   = lane & 15;
  const int lh   = lane >> 4;

  // ---- Phase A: gather Q/Aw rows of layer1, convert bf16, write LQ/LA ----
  {
    const int r   = t & 127;
    const int isA = t >> 7;
    const int node = isA ? aidx[b * 128 + r] : qidx[b * 128 + r];
    const float* src = layer1 + (size_t)node * 256;
    unsigned short* dst = (isA ? R2u : R1u) + r * 264;
#pragma unroll 8
    for (int c = 0; c < 256; c += 8) {
      f32x4 f0 = *(const f32x4*)(src + c);
      f32x4 f1 = *(const f32x4*)(src + c + 4);
      PU8 p;
      p.w[0] = (unsigned int)f2bf(f0[0]) | ((unsigned int)f2bf(f0[1]) << 16);
      p.w[1] = (unsigned int)f2bf(f0[2]) | ((unsigned int)f2bf(f0[3]) << 16);
      p.w[2] = (unsigned int)f2bf(f1[0]) | ((unsigned int)f2bf(f1[1]) << 16);
      p.w[3] = (unsigned int)f2bf(f1[2]) | ((unsigned int)f2bf(f1[3]) << 16);
      *(u16x8*)(dst + c) = p.v;
    }
  }
  __syncthreads();

  // ---- Phase B: S = Q @ Aw^T  (wave w owns row-tile w; K = 256) ----
  {
    const int i0 = w * 16;
    v8f acc[8];
#pragma unroll
    for (int i = 0; i < 8; ++i) acc[i] = v8zero();
#pragma unroll
    for (int kk = 0; kk < 8; ++kk) {
      const int k0 = kk * 32;
      u16x8 a0 = *(const u16x8*)(&R1u[(i0 + ln) * 264 + k0 + 16 * lh]);
      u16x8 a1 = *(const u16x8*)(&R1u[(i0 + ln) * 264 + k0 + 16 * lh + 8]);
      u16x16 a;
#pragma unroll
      for (int i = 0; i < 8; ++i) { a[i] = a0[i]; a[i + 8] = a1[i]; }
#pragma unroll
      for (int tj = 0; tj < 8; ++tj) {
        u16x16 bb = *(const u16x16*)(&R2u[(tj * 16 + ln) * 264 + k0 + 16 * lh]);
        acc[tj] = wmma_bf16(a, bb, acc[tj]);
      }
    }
#pragma unroll
    for (int tj = 0; tj < 8; ++tj)
#pragma unroll
      for (int v = 0; v < 8; ++v)
        S[(i0 + v + 8 * lh) * 132 + tj * 16 + ln] = acc[tj][v];
  }
  __syncthreads();

  // ---- Phase C: row / column softmax statistics ----
  {
    if (t < 128) {
      const float* sr = S + t * 132;
      float mx = -3.4e38f;
      for (int j = 0; j < 128; ++j) mx = fmaxf(mx, sr[j]);
      float sm = 0.f;
      for (int j = 0; j < 128; ++j) sm += __expf(sr[j] - mx);
      rowmax[t] = mx; rowsum[t] = sm;
    } else {
      const int c = t - 128;
      float mx = -3.4e38f;
      for (int i = 0; i < 128; ++i) mx = fmaxf(mx, S[i * 132 + c]);
      float sm = 0.f;
      for (int i = 0; i < 128; ++i) sm += __expf(S[i * 132 + c] - mx);
      colmax[c] = mx; colsum[c] = sm;
    }
  }
  __syncthreads();

  // ---- Phase C2: in-place transpose LQ->LQT, LA->LAT (register-staged) ----
  {
    unsigned int tq[64];
#pragma unroll
    for (int i = 0; i < 128; i += 2)
      tq[i >> 1] = (unsigned int)R1u[i * 264 + t] |
                   ((unsigned int)R1u[(i + 1) * 264 + t] << 16);
    __syncthreads();
#pragma unroll
    for (int c = 0; c < 8; ++c) {
      PU16 p;
#pragma unroll
      for (int i = 0; i < 8; ++i) p.w[i] = tq[c * 8 + i];
      *(u16x16*)(&R1u[t * 136 + c * 16]) = p.v;
    }
    __syncthreads();
#pragma unroll
    for (int i = 0; i < 128; i += 2)
      tq[i >> 1] = (unsigned int)R2u[i * 264 + t] |
                   ((unsigned int)R2u[(i + 1) * 264 + t] << 16);
    __syncthreads();
#pragma unroll
    for (int c = 0; c < 8; ++c) {
      PU16 p;
#pragma unroll
      for (int i = 0; i < 8; ++i) p.w[i] = tq[c * 8 + i];
      *(u16x16*)(&R2u[t * 136 + c * 16]) = p.v;
    }
    __syncthreads();
  }

  // ---- Phase D: S -> softmax weights WQ[i][j], WAT[j][i] (overlays S) ----
  {
    const int r = t >> 1, j0 = (t & 1) * 64;
    float sreg[64];
#pragma unroll
    for (int j = 0; j < 64; ++j) sreg[j] = S[r * 132 + j0 + j];
    __syncthreads();
    const float rm = rowmax[r];
    const float rsi = 1.f / rowsum[r];
#pragma unroll
    for (int j = 0; j < 64; ++j) {
      const int jj = j0 + j;
      WQp[r * 136 + jj]  = f2bf(__expf(sreg[j] - rm) * rsi);
      WATp[jj * 136 + r] = f2bf(__expf(sreg[j] - colmax[jj]) / colsum[jj]);
    }
  }
  __syncthreads();

  // ---- Phase E: EQ = WQ@Aw, EA = WAT@Q, fused reductions -> logits ----
  {
    const float invL = 0.08838834764831845f;   // 1/sqrt(128)
    float l0 = 0.f, l1 = 0.f;
#pragma unroll
    for (int side = 0; side < 2; ++side) {
      const unsigned short* Am = side ? WATp : WQp;   // A operand (weights)
      const unsigned short* Bm = side ? R1u : R2u;    // B operand: LQT / LAT
      const unsigned short* Em = side ? R2u : R1u;    // elementwise: LAT / LQT
      const int wr = side ? 256 : 0;                  // qr / ar slot in W
      const int wm = side ? 768 : 512;                // mul_q / mul_a slot
#pragma unroll
      for (int ss = 0; ss < 2; ++ss) {
        const int h0 = (w + 8 * ss) * 16;
        float racc = 0.f, macc = 0.f;
#pragma unroll
        for (int ti = 0; ti < 8; ++ti) {
          const int i0 = ti * 16;
          v8f acc = v8zero();
#pragma unroll
          for (int kk = 0; kk < 4; ++kk) {
            const int k0 = kk * 32;
            u16x8 a0 = *(const u16x8*)(&Am[(i0 + ln) * 136 + k0 + 16 * lh]);
            u16x8 a1 = *(const u16x8*)(&Am[(i0 + ln) * 136 + k0 + 16 * lh + 8]);
            u16x16 a;
#pragma unroll
            for (int i = 0; i < 8; ++i) { a[i] = a0[i]; a[i + 8] = a1[i]; }
            u16x16 bb = *(const u16x16*)(&Bm[(h0 + ln) * 136 + k0 + 16 * lh]);
            acc = wmma_bf16(a, bb, acc);
          }
#pragma unroll
          for (int v = 0; v < 8; ++v) {
            const float qv = bf2f(Em[(h0 + ln) * 136 + i0 + v + 8 * lh]);
            const float ev = acc[v];
            const float d = qv - ev;
            racc += d * d;
            macc += qv * ev;
          }
        }
        const int hcol = h0 + ln;
        l0 += invL * (racc * Wmat[wr + hcol]        + macc * Wmat[wm + hcol]);
        l1 += invL * (racc * Wmat[1024 + wr + hcol] + macc * Wmat[1024 + wm + hcol]);
      }
    }
    contrib[t * 2]     = l0;
    contrib[t * 2 + 1] = l1;
  }
  __syncthreads();

  if (t == 0) {
    float l0 = bias[0], l1 = bias[1];
    for (int i = 0; i < 256; ++i) { l0 += contrib[i * 2]; l1 += contrib[i * 2 + 1]; }
    const float mx = fmaxf(l0, l1);
    const float lse = mx + logf(expf(l0 - mx) + expf(l1 - mx));
    out[b * 2]     = l0 - lse;
    out[b * 2 + 1] = l1 - lse;
  }
}

// ---------------------------------------------------------------------------
extern "C" void kernel_launch(void* const* d_in, const int* in_sizes, int n_in,
                              void* d_out, int out_size, void* d_ws, size_t ws_size,
                              hipStream_t stream) {
  const float* A    = (const float*)d_in[0];   // [8192,8192]
  const float* E    = (const float*)d_in[1];   // [8192,256]
  const float* W    = (const float*)d_in[2];   // [2,1024]
  const float* bias = (const float*)d_in[3];   // [2]
  const int*   qi   = (const int*)d_in[4];     // [256,128]
  const int*   ai   = (const int*)d_in[5];     // [256,128]

  // workspace carve: layer1 (8MB f32) | Et[2][256][8192] u16 (hi, lo planes)
  float*          layer1 = (float*)d_ws;
  unsigned short* Et     = (unsigned short*)((char*)d_ws + (size_t)8192 * 256 * 4);
  unsigned short* Etlo   = Et + (size_t)8192 * 256;

  k_prep_E<<<dim3(1024), dim3(256), 0, stream>>>(E, Et, Etlo);
  k_gemm_layer1<<<dim3(128), dim3(128), K1_LDS_BYTES, stream>>>(A, Et, layer1);
  k_attn<<<dim3(256), dim3(256), SM_TOTAL, stream>>>(layer1, W, bias, qi, ai,
                                                     (float*)d_out);
}
